// BinaryGapLoss_7868380086298
// MI455X (gfx1250) — matrix operand order repaired and checked by reading, not compiled, not run on hardware
//
#include <hip/hip_runtime.h>
#include <stdint.h>
#include <math.h>

// ---------------------------------------------------------------------------
// BinaryGapLoss for MI455X (gfx1250, wave32, WMMA, 320KB LDS/WGP)
//
// Pipeline (all on `stream`, fixed work -> graph-capture safe):
//   k_init        : zero the loss accumulator
//   k_binarize    : pred >= 0.5 -> 1-bit/pixel bitmap via wave32 ballot
//   k_thin        : Zhang-Suen thinning, bit-sliced, fully LDS-resident
//                   (2 x 1024x33 word ping-pong buffers ~ 264KB of the
//                   CDNA5 320KB LDS), <=64 iterations w/ convergence flag.
//                   Global<->LDS staging via GLOBAL_*_ASYNC_*_LDS_B128 DMA
//                   (ASYNCcnt) instead of VGPR round trips.
//   k_endpoints   : skeleton pixels with exactly 1 neighbor (bit-sliced CSA)
//   k_hconv       : horizontal 9-tap box sum as GEMM: tile(16x24) x band(24x16)
//                   via v_wmma_f32_16x16x32_f16  -> f16 intermediate
//   k_vconv_loss  : vertical 9-tap as GEMM: band(16x24) x H(24x16) via WMMA,
//                   fused with BCE + weight + wave-reduced atomic accumulate
//   k_final       : mean
// ---------------------------------------------------------------------------

typedef __attribute__((ext_vector_type(16))) _Float16 v16h;
typedef __attribute__((ext_vector_type(8)))  float    v8f;

#define IMGS 8
#define IMH 1024
#define IMW 1024
#define WPR 32                       // 32-bit words per row
#define WORDS_PER_IMG (IMH * WPR)    // 32768 words = 128KB/image
#define PIX_PER_IMG   (IMH * IMW)
#define PIX_TOTAL     (IMGS * PIX_PER_IMG)
#define TILES_1D      (IMW / 16)     // 64
#define TILES_PER_IMG (TILES_1D * TILES_1D)
#define TOTAL_TILES   (IMGS * TILES_PER_IMG)   // 32768 tiles (1 wave each)
#define K_WEIGHT 60.0f

// workspace layout (bytes)
#define WS_SKEL 0u                   // 1MB: threshold bitmap -> skeleton (in place)
#define WS_ENDP (1u << 20)           // 1MB: endpoint bitmap
#define WS_HBUF (2u << 20)           // 16MB: f16 horizontal box sums
#define WS_SUM  (18u << 20)          // accumulator

#define LDSPITCH 33                  // 33-word row pitch -> LDS bank-conflict free

// ---------------- bit-sliced helpers ----------------------------------------

__device__ __forceinline__ void csa(uint32_t a, uint32_t b, uint32_t c,
                                    uint32_t& s, uint32_t& cy) {
  uint32_t ab = a ^ b;
  s  = ab ^ c;
  cy = (a & b) | (ab & c);
}

// Sum of 8 one-bit-per-lane vectors -> bit-sliced 4-bit count {b0..b3}
__device__ __forceinline__ void sum8(uint32_t x0, uint32_t x1, uint32_t x2,
                                     uint32_t x3, uint32_t x4, uint32_t x5,
                                     uint32_t x6, uint32_t x7,
                                     uint32_t& b0, uint32_t& b1,
                                     uint32_t& b2, uint32_t& b3) {
  uint32_t s0, c0, s1, c1, s2, c2, s3, c3, s4, c4, s5, c5, s6, c6;
  csa(x0, x1, x2, s0, c0);
  csa(x3, x4, x5, s1, c1);
  csa(x6, x7, 0u, s2, c2);
  csa(s0, s1, s2, s3, c3);          // s3 = ones place
  csa(c0, c1, c2, s4, c4);          // twos-place partials
  csa(s4, c3, 0u, s5, c5);          // s5 = twos place
  csa(c4, c5, 0u, s6, c6);          // s6 = fours, c6 = eights
  b0 = s3; b1 = s5; b2 = s6; b3 = c6;
}

// 8 neighbors (P2..P9 = N,NE,E,SE,S,SW,W,NW), bit i = pixel w*32+i
__device__ __forceinline__ void neigh8(uint32_t nL, uint32_t nC, uint32_t nR,
                                       uint32_t cL, uint32_t cC, uint32_t cR,
                                       uint32_t sL, uint32_t sC, uint32_t sR,
                                       uint32_t P[8]) {
  P[0] = nC;                          // P2 N
  P[1] = (nC >> 1) | (nR << 31);      // P3 NE  (pixel x+1 -> bit i+1)
  P[2] = (cC >> 1) | (cR << 31);      // P4 E
  P[3] = (sC >> 1) | (sR << 31);      // P5 SE
  P[4] = sC;                          // P6 S
  P[5] = (sC << 1) | (sL >> 31);      // P7 SW
  P[6] = (cC << 1) | (cL >> 31);      // P8 W
  P[7] = (nC << 1) | (nL >> 31);      // P9 NW
}

// ---------------- k_init / k_final ------------------------------------------

__global__ void k_init(float* sum) { *sum = 0.0f; }

__global__ void k_final(const float* sum, float* out) {
  out[0] = sum[0] * (1.0f / (float)PIX_TOTAL);
}

// ---------------- k_binarize: ballot threshold ------------------------------

__global__ __launch_bounds__(256) void k_binarize(const float* __restrict__ pred,
                                                  uint32_t* __restrict__ bm) {
  uint32_t gid = blockIdx.x * blockDim.x + threadIdx.x;
  float p = pred[gid];
  uint32_t mask = __builtin_amdgcn_ballot_w32(p >= 0.5f);   // wave32 ballot
  if ((threadIdx.x & 31u) == 0u) bm[gid >> 5] = mask;
}

// ---------------- k_thin: LDS-resident Zhang-Suen ---------------------------

__device__ __forceinline__ int thin_row(const uint32_t* __restrict__ src,
                                        uint32_t* __restrict__ dst,
                                        int r, int sub) {
  const uint32_t* rowC = src + r * LDSPITCH;
  const uint32_t* rowN = (r > 0)         ? src + (r - 1) * LDSPITCH : nullptr;
  const uint32_t* rowS = (r < IMH - 1)   ? src + (r + 1) * LDSPITCH : nullptr;
  uint32_t* rowD = dst + r * LDSPITCH;
  int changed = 0;
#pragma unroll 4
  for (int w = 0; w < WPR; ++w) {
    uint32_t cC = rowC[w];
    uint32_t cL = (w > 0)       ? rowC[w - 1] : 0u;
    uint32_t cR = (w < WPR - 1) ? rowC[w + 1] : 0u;
    uint32_t nC = rowN ? rowN[w] : 0u;
    uint32_t nL = (rowN && w > 0)       ? rowN[w - 1] : 0u;
    uint32_t nR = (rowN && w < WPR - 1) ? rowN[w + 1] : 0u;
    uint32_t sC = rowS ? rowS[w] : 0u;
    uint32_t sL = (rowS && w > 0)       ? rowS[w - 1] : 0u;
    uint32_t sR = (rowS && w < WPR - 1) ? rowS[w + 1] : 0u;

    uint32_t P[8];
    neigh8(nL, nC, nR, cL, cC, cR, sL, sC, sR, P);

    // B = neighbor count; need 2 <= B <= 6
    uint32_t b0, b1, b2, b3;
    sum8(P[0], P[1], P[2], P[3], P[4], P[5], P[6], P[7], b0, b1, b2, b3);
    uint32_t condB = (b1 | b2 | b3) & ~(b3 | (b2 & b1 & b0));

    // A = number of 0->1 transitions in P2,P3,...,P9,P2; need A == 1
    uint32_t t0 = ~P[0] & P[1], t1 = ~P[1] & P[2], t2 = ~P[2] & P[3],
             t3 = ~P[3] & P[4], t4 = ~P[4] & P[5], t5 = ~P[5] & P[6],
             t6 = ~P[6] & P[7], t7 = ~P[7] & P[0];
    uint32_t a0, a1, a2, a3;
    sum8(t0, t1, t2, t3, t4, t5, t6, t7, a0, a1, a2, a3);
    uint32_t condA = a0 & ~a1 & ~a2 & ~a3;

    uint32_t c3m, c4m;
    if (sub == 0) { c3m = ~(P[0] & P[2] & P[4]); c4m = ~(P[2] & P[4] & P[6]); }
    else          { c3m = ~(P[0] & P[2] & P[6]); c4m = ~(P[0] & P[4] & P[6]); }

    uint32_t rm = cC & condB & condA & c3m & c4m;
    uint32_t nw = cC & ~rm;
    rowD[w] = nw;
    changed |= (nw != cC);
  }
  return changed;
}

__global__ __launch_bounds__(1024) void k_thin(uint32_t* __restrict__ bm) {
  extern __shared__ uint32_t smem[];
  uint32_t* buf0 = smem;                         // 1024*33 words (dyn-LDS base 0)
  uint32_t* buf1 = smem + IMH * LDSPITCH;        // 1024*33 words
  int* flag = (int*)(smem + 2 * IMH * LDSPITCH);

  const int tid = threadIdx.x;
  uint32_t* g = bm + (size_t)blockIdx.x * WORDS_PER_IMG;

  // ---- async DMA: packed global -> padded LDS (ASYNCcnt path, no VGPR trip)
  // chunk i covers words [4i, 4i+4); rows are 32 words so no row straddle.
#pragma unroll
  for (int k = 0; k < 8; ++k) {
    int i = tid + k * 1024;                          // chunk index 0..8191
    int r4 = i >> 3;                                 // row
    int w4 = (i & 7) << 2;                           // word within row
    uint32_t ldsoff = (uint32_t)((r4 * LDSPITCH + w4) << 2);  // byte offset in buf0
    const uint32_t* gp = g + ((size_t)i << 2);
    asm volatile("global_load_async_to_lds_b128 %0, %1, off"
                 :: "v"(ldsoff), "v"(gp) : "memory");
  }
  asm volatile("s_wait_asynccnt 0" ::: "memory");
  __syncthreads();

  const int r = tid;   // one row per thread
  for (int iter = 0; iter < 64; ++iter) {
    if (tid == 0) *flag = 0;
    __syncthreads();
    int ch = thin_row(buf0, buf1, r, 0);
    __syncthreads();
    ch |= thin_row(buf1, buf0, r, 1);
    if (ch) atomicOr(flag, 1);
    __syncthreads();
    int any = *flag;
    __syncthreads();
    if (!any) break;                 // uniform across block
  }

  // ---- async DMA: padded LDS -> packed global (skeleton, in place)
#pragma unroll
  for (int k = 0; k < 8; ++k) {
    int i = tid + k * 1024;
    int r4 = i >> 3;
    int w4 = (i & 7) << 2;
    uint32_t ldsoff = (uint32_t)((r4 * LDSPITCH + w4) << 2);
    uint32_t* gp = g + ((size_t)i << 2);
    asm volatile("global_store_async_from_lds_b128 %0, %1, off"
                 :: "v"(gp), "v"(ldsoff) : "memory");
  }
  asm volatile("s_wait_asynccnt 0" ::: "memory");
}

// ---------------- k_endpoints: skeleton pixels with exactly 1 neighbor ------

__global__ __launch_bounds__(256) void k_endpoints(const uint32_t* __restrict__ sk,
                                                   uint32_t* __restrict__ ep) {
  uint32_t idx = blockIdx.x * blockDim.x + threadIdx.x;   // word index
  uint32_t im = idx >> 15;
  uint32_t rw = idx & 32767u;
  int r = (int)(rw >> 5);
  int w = (int)(rw & 31u);
  const uint32_t* g = sk + (size_t)im * WORDS_PER_IMG;

  uint32_t cC = g[r * WPR + w];
  uint32_t cL = (w > 0)       ? g[r * WPR + w - 1] : 0u;
  uint32_t cR = (w < WPR - 1) ? g[r * WPR + w + 1] : 0u;
  uint32_t nC = (r > 0)       ? g[(r - 1) * WPR + w] : 0u;
  uint32_t nL = (r > 0 && w > 0)       ? g[(r - 1) * WPR + w - 1] : 0u;
  uint32_t nR = (r > 0 && w < WPR - 1) ? g[(r - 1) * WPR + w + 1] : 0u;
  uint32_t sC = (r < IMH - 1) ? g[(r + 1) * WPR + w] : 0u;
  uint32_t sL = (r < IMH - 1 && w > 0)       ? g[(r + 1) * WPR + w - 1] : 0u;
  uint32_t sR = (r < IMH - 1 && w < WPR - 1) ? g[(r + 1) * WPR + w + 1] : 0u;

  uint32_t P[8];
  neigh8(nL, nC, nR, cL, cC, cR, sL, sC, sR, P);
  uint32_t b0, b1, b2, b3;
  sum8(P[0], P[1], P[2], P[3], P[4], P[5], P[6], P[7], b0, b1, b2, b3);
  // exactly one neighbor, and pixel on skeleton
  ep[idx] = cC & b0 & ~b1 & ~b2 & ~b3;
}

// ---------------- WMMA operand mapping --------------------------------------
// f16 A-matrix 16x32 (ISA 7.12.2): lane L holds row M=L%16, half=L/16;
// element e (of 16 per lane) -> K = 16*(e/8) + 8*half + 2*((e/2)%4) + (e%2)
__device__ __forceinline__ int a_kslot(int e, int half) {
  return ((e >> 3) << 4) + (half << 3) + (((e >> 1) & 3) << 1) + (e & 1);
}

// ---------------- k_hconv: horizontal 9-tap via WMMA ------------------------
// H[i][j] = sum_{d=-4..4} C[r0+i][c0+j+d]   as  A(16x32 img) x B(32x16 band)

__global__ __launch_bounds__(256) void k_hconv(const uint32_t* __restrict__ ep,
                                               _Float16* __restrict__ Hbuf) {
  int wave = (int)((blockIdx.x * blockDim.x + threadIdx.x) >> 5);
  int lane = (int)(threadIdx.x & 31u);
  int im = wave >> 12;
  int t  = wave & 4095;
  int r0 = (t >> 6) << 4;
  int c0 = (t & 63) << 4;

  int M = lane & 15, half = lane >> 4;
  int row = r0 + M;
  const uint32_t* g = ep + (size_t)im * WORDS_PER_IMG + (size_t)row * WPR;

  // A: image slab rows r0..r0+15, K-cols map to image cols c0-4+K (K<24 live)
  v16h a;
#pragma unroll
  for (int e = 0; e < 16; ++e) {
    int K = a_kslot(e, half);
    int col = c0 - 4 + K;
    uint32_t bit = 0u;
    if (K < 24 && col >= 0 && col < IMW) bit = (g[col >> 5] >> (col & 31)) & 1u;
    a[e] = (_Float16)(float)bit;
  }
  // B: band, column j = lane%16, row k = a_kslot(e, half); B[k][j]=1 iff j<=k<=j+8
  v16h b;
  int nB = lane & 15;
#pragma unroll
  for (int e = 0; e < 16; ++e) {
    int k = a_kslot(e, half);
    b[e] = (k < 24 && k >= nB && k <= nB + 8) ? (_Float16)1.0f : (_Float16)0.0f;
  }

  v8f c = {};
  c = __builtin_amdgcn_wmma_f32_16x16x32_f16(false, a, false, b,
                                             (short)0, c, false, false);

  // D layout: lane -> N=lane%16 col, VGPR v -> M = v + 8*(lane/16)
  _Float16* Hrow = Hbuf + (size_t)im * PIX_PER_IMG;
  int colD = c0 + (lane & 15);
#pragma unroll
  for (int v = 0; v < 8; ++v) {
    int rowD = r0 + v + 8 * half;
    Hrow[(size_t)rowD * IMW + colD] = (_Float16)c[v];
  }
}

// ---------------- k_vconv_loss: vertical 9-tap via WMMA + fused BCE ---------
// N[i][j] = sum_{d=-4..4} H[r0+i+d][c0+j]  as  A(16x32 band) x B(32x16 H-slab)

__global__ __launch_bounds__(256) void k_vconv_loss(const _Float16* __restrict__ Hbuf,
                                                    const float* __restrict__ pred,
                                                    const float* __restrict__ target,
                                                    float* __restrict__ sum) {
  int wave = (int)((blockIdx.x * blockDim.x + threadIdx.x) >> 5);
  int lane = (int)(threadIdx.x & 31u);
  int im = wave >> 12;
  int t  = wave & 4095;
  int r0 = (t >> 6) << 4;
  int c0 = (t & 63) << 4;

  int M = lane & 15, half = lane >> 4;

  // B: column = c0 + lane%16, element e -> row k = a_kslot(e, half) -> image row r0-4+k
  v16h b;
#pragma unroll
  for (int e = 0; e < 16; ++e) {
    int r = r0 - 4 + a_kslot(e, half);
    b[e] = (r >= 0 && r < IMH) ? Hbuf[(size_t)im * PIX_PER_IMG + (size_t)r * IMW + c0 + (lane & 15)] : (_Float16)0.0f;
  }
  // A: band, A[M][K] = 1 iff M <= K <= M+8 (K<24 by construction)
  v16h a;
#pragma unroll
  for (int e = 0; e < 16; ++e) {
    int K = a_kslot(e, half);
    a[e] = (K >= M && K <= M + 8) ? (_Float16)1.0f : (_Float16)0.0f;
  }

  v8f c = {};
  c = __builtin_amdgcn_wmma_f32_16x16x32_f16(false, a, false, b,
                                             (short)0, c, false, false);

  const float* P = pred   + (size_t)im * PIX_PER_IMG;
  const float* T = target + (size_t)im * PIX_PER_IMG;
  int colD = c0 + (lane & 15);
  __builtin_prefetch(P + (size_t)r0 * IMW + colD, 0, 0);   // global_prefetch_b8
  __builtin_prefetch(T + (size_t)r0 * IMW + colD, 0, 0);

  float acc = 0.0f;
#pragma unroll
  for (int v = 0; v < 8; ++v) {
    int rowD = r0 + v + 8 * half;
    size_t idx = (size_t)rowD * IMW + colD;
    float p = P[idx];
    float tt = T[idx];
    float n = c[v];                               // 9x9 box sum of endpoints
    float wgt = (n > 0.0f) ? K_WEIGHT * n : 1.0f; // W = K*N + (N==0)
    float l = -(tt * logf(p) + (1.0f - tt) * log1pf(-p));
    acc += wgt * l;
  }
  // wave32 reduction
#pragma unroll
  for (int o = 16; o > 0; o >>= 1) acc += __shfl_down(acc, o, 32);
  if (lane == 0) atomicAdd(sum, acc);
}

// ---------------------------------------------------------------------------

extern "C" void kernel_launch(void* const* d_in, const int* in_sizes, int n_in,
                              void* d_out, int out_size, void* d_ws, size_t ws_size,
                              hipStream_t stream) {
  const float* pred   = (const float*)d_in[0];
  const float* target = (const float*)d_in[1];
  float* out = (float*)d_out;

  uint8_t* ws = (uint8_t*)d_ws;
  uint32_t* skel  = (uint32_t*)(ws + WS_SKEL);
  uint32_t* ep    = (uint32_t*)(ws + WS_ENDP);
  _Float16* Hbuf  = (_Float16*)(ws + WS_HBUF);
  float*    sum   = (float*)(ws + WS_SUM);

  k_init<<<1, 1, 0, stream>>>(sum);

  k_binarize<<<PIX_TOTAL / 256, 256, 0, stream>>>(pred, skel);

  // 2 x (1024x33) ping-pong bitmaps + flag, ~264KB of the 320KB CDNA5 LDS
  size_t smem = (size_t)(2 * IMH * LDSPITCH + 16) * sizeof(uint32_t);
  k_thin<<<IMGS, 1024, smem, stream>>>(skel);

  k_endpoints<<<(IMGS * WORDS_PER_IMG) / 256, 256, 0, stream>>>(skel, ep);

  k_hconv<<<TOTAL_TILES / 8, 256, 0, stream>>>(ep, Hbuf);

  k_vconv_loss<<<TOTAL_TILES / 8, 256, 0, stream>>>(Hbuf, pred, target, sum);

  k_final<<<1, 1, 0, stream>>>(sum, out);
}
